// EdgeDecoder_19859928777345
// MI455X (gfx1250) — matrix operand order repaired and checked by reading, not compiled, not run on hardware
//
#include <hip/hip_runtime.h>
#include <hip/hip_bf16.h>

typedef __attribute__((ext_vector_type(16))) __bf16 v16bf;
typedef __attribute__((ext_vector_type(8)))  float  v8f;
typedef __attribute__((ext_vector_type(4)))  int    v4i;

union Frag { v16bf v; uint4 u[2]; };

#define HID 128
#define LDA 264   // 256 + 8 halves padding -> conflict-free b128 fragment reads

#if defined(__has_builtin)
# if __has_builtin(__builtin_amdgcn_global_load_async_to_lds_b128) && \
     __has_builtin(__builtin_amdgcn_s_wait_asynccnt)
#  define HAVE_ASYNC_LDS 1
# endif
#endif
#ifndef HAVE_ASYNC_LDS
# define HAVE_ASYNC_LDS 0
#endif

#if HAVE_ASYNC_LDS
typedef __attribute__((address_space(1))) v4i* gv4i_p;   // global int4 vector ptr
typedef __attribute__((address_space(3))) v4i* lv4i_p;   // LDS int4 vector ptr
#endif

static __device__ __forceinline__ unsigned f2bf(float x) {
    union { float f; unsigned u; } c; c.f = x;
    return (c.u + 0x7FFFu + ((c.u >> 16) & 1u)) >> 16;   // RNE
}

// ---- DPP 16-lane sum: XOR butterfly, basis {1,2,4,15} (15 == row_mirror) ----
template<int M>
static __device__ __forceinline__ float xor_add8(float v) {
    constexpr int sel = ((0^M)&7)        | (((1^M)&7)<<3)  | (((2^M)&7)<<6)  | (((3^M)&7)<<9)
                      | (((4^M)&7)<<12)  | (((5^M)&7)<<15) | (((6^M)&7)<<18) | (((7^M)&7)<<21);
    int t = __builtin_amdgcn_mov_dpp8(__builtin_bit_cast(int, v), sel);
    return v + __builtin_bit_cast(float, t);
}
static __device__ __forceinline__ float mirror_add16(float v) {
    int t = __builtin_amdgcn_mov_dpp(__builtin_bit_cast(int, v), 0x140 /*row_mirror*/,
                                     0xf, 0xf, true);
    return v + __builtin_bit_cast(float, t);
}
static __device__ __forceinline__ float reduce16(float v) {
    v = xor_add8<1>(v); v = xor_add8<2>(v); v = xor_add8<4>(v);
    return mirror_add16(v);
}

// W1 [256,128] f32 (K-major) -> w1t [128][256] bf16 (N rows, K contiguous)
__global__ void prep_w1t_kernel(const float* __restrict__ w1,
                                unsigned short* __restrict__ w1t) {
    int i = blockIdx.x * blockDim.x + threadIdx.x;
    if (i < HID * 2 * HID) {
        int n = i >> 8, k = i & 255;
        w1t[n * 256 + k] = (unsigned short)f2bf(w1[k * HID + n]);
    }
}

// z table f32 -> bf16, 4 elements per thread
__global__ void cvt_bf16_kernel(const float* __restrict__ src,
                                unsigned short* __restrict__ dst, int n4) {
    int i = blockIdx.x * blockDim.x + threadIdx.x;
    if (i < n4) {
        float4 v = ((const float4*)src)[i];
        uint2 p;
        p.x = f2bf(v.x) | (f2bf(v.y) << 16);
        p.y = f2bf(v.z) | (f2bf(v.w) << 16);
        ((uint2*)dst)[i] = p;
    }
}

__global__ __launch_bounds__(128) void edge_mlp_kernel(
    const float* __restrict__ z_p, const float* __restrict__ z_o,
    const unsigned short* __restrict__ zp_bf, const unsigned short* __restrict__ zo_bf,
    const int* __restrict__ idx0, const int* __restrict__ idx1,
    const int* __restrict__ idx2,
    const unsigned short* __restrict__ w1t,
    const float* __restrict__ b1, const float* __restrict__ w2,
    const float* __restrict__ b2,
    float* __restrict__ out, int E, int numTiles, int use_bf16)
{
    __shared__ __align__(16) unsigned short sA[2][32 * LDA];   // double-buffered tiles
    __shared__ float sOut[2][32];

    const int tid   = threadIdx.x;
    const int lane  = tid & 31;
    const int wv    = tid >> 5;
    const int nhalf = lane >> 4;
    const int nlo   = lane & 15;
    const int total = 3 * E;

    // ---- preload B fragments (2 N-tiles x 8 K-steps) into VGPRs for the whole run ----
    Frag B0[8], B1[8];
    {
        const unsigned short* p0 = w1t + (size_t)(16 * (2 * wv)     + nlo) * 256 + nhalf * 16;
        const unsigned short* p1 = w1t + (size_t)(16 * (2 * wv + 1) + nlo) * 256 + nhalf * 16;
#pragma unroll
        for (int k = 0; k < 8; ++k) {
            B0[k].u[0] = *(const uint4*)(p0 + 32 * k);
            B0[k].u[1] = *(const uint4*)(p0 + 32 * k + 8);
            B1[k].u[0] = *(const uint4*)(p1 + 32 * k);
            B1[k].u[1] = *(const uint4*)(p1 + 32 * k + 8);
        }
    }
    const float b1v0 = b1[16 * (2 * wv)     + nlo];
    const float b1v1 = b1[16 * (2 * wv + 1) + nlo];
    const float w2v0 = w2[16 * (2 * wv)     + nlo];
    const float w2v1 = w2[16 * (2 * wv + 1) + nlo];
    const float b2v  = b2[0];

    // Issue the gather for `tile` into buffer `buf`. Always exactly 8 async ops
    // per thread in the bf16 path (tail edges are clamped; outputs guarded later).
    auto gather = [&](int tile, int buf) {
        const int er  = tid >> 2;        // edge within tile (0..31)
        const int seg = tid & 3;         // 64-feature segment
        int e = tile * 32 + er;
        if (e >= total) e = total - 1;   // clamp: keeps async-op count uniform
        int el; const int* idx;
        const float *sf, *df; const unsigned short *sh, *dh;
        if (e < E)          { el = e;         idx = idx0; sf = z_p; df = z_o; sh = zp_bf; dh = zo_bf; }
        else if (e < 2 * E) { el = e - E;     idx = idx1; sf = z_o; df = z_p; sh = zo_bf; dh = zp_bf; }
        else                { el = e - 2 * E; idx = idx2; sf = z_o; df = z_o; sh = zo_bf; dh = zo_bf; }
        const int f0 = seg * 64;
        int row, col;
        const float* tf; const unsigned short* th;
        if (f0 < HID) { row = idx[el];     col = f0;       tf = sf; th = sh; }
        else          { row = idx[E + el]; col = f0 - HID; tf = df; th = dh; }
        unsigned short* dst = &sA[buf][er * LDA + seg * 64];
        if (use_bf16) {
            const unsigned short* g = th + (size_t)row * HID + col;
#if HAVE_ASYNC_LDS
            gv4i_p gp = (gv4i_p)(void*)const_cast<unsigned short*>(g);
            lv4i_p lp = (lv4i_p)(void*)dst;
#pragma unroll
            for (int j = 0; j < 8; ++j)
                __builtin_amdgcn_global_load_async_to_lds_b128(gp + j, lp + j, 0, 0);
#else
            const uint4* p = (const uint4*)g;
#pragma unroll
            for (int j = 0; j < 8; ++j) *(uint4*)(dst + 8 * j) = p[j];
#endif
        } else {
            const float4* p = (const float4*)(tf + (size_t)row * HID + col);
#pragma unroll
            for (int j = 0; j < 8; ++j) {
                float4 a = p[2 * j], b = p[2 * j + 1];
                uint4 pk;
                pk.x = f2bf(a.x) | (f2bf(a.y) << 16);
                pk.y = f2bf(a.z) | (f2bf(a.w) << 16);
                pk.z = f2bf(b.x) | (f2bf(b.y) << 16);
                pk.w = f2bf(b.z) | (f2bf(b.w) << 16);
                *(uint4*)(dst + 8 * j) = pk;
            }
        }
    };

    int bi = 0;
    int tile = blockIdx.x;
    if (tile < numTiles) gather(tile, 0);
    if (tid < 32) sOut[0][tid] = 0.f;

    for (; tile < numTiles; tile += gridDim.x) {
        const int nxt = tile + gridDim.x;
        // Prefetch next tile into the other buffer while this one is consumed.
        if (nxt < numTiles) gather(nxt, bi ^ 1);
#if HAVE_ASYNC_LDS
        if (nxt < numTiles) __builtin_amdgcn_s_wait_asynccnt(8);   // drain older (current-tile) 8
        else                __builtin_amdgcn_s_wait_asynccnt(0);   // drain everything
#endif
        __syncthreads();
        if (tid < 32) sOut[bi ^ 1][tid] = 0.f;   // prep accumulator for next iteration

        // ---- GEMM: two 16-row M-tiles x two 16-col N-tiles, K=256 in 8 steps ----
        v8f acc00 = {}, acc01 = {}, acc10 = {}, acc11 = {};
        {
            const unsigned short* a0 = &sA[bi][nlo * LDA        + nhalf * 16];
            const unsigned short* a1 = &sA[bi][(16 + nlo) * LDA + nhalf * 16];
#pragma unroll
            for (int k = 0; k < 8; ++k) {
                Frag A0, A1;
                A0.u[0] = *(const uint4*)(a0 + 32 * k);
                A0.u[1] = *(const uint4*)(a0 + 32 * k + 8);
                A1.u[0] = *(const uint4*)(a1 + 32 * k);
                A1.u[1] = *(const uint4*)(a1 + 32 * k + 8);
                acc00 = __builtin_amdgcn_wmma_f32_16x16x32_bf16(
                    false, A0.v, false, B0[k].v, (short)0, acc00, false, false);
                acc01 = __builtin_amdgcn_wmma_f32_16x16x32_bf16(
                    false, A0.v, false, B1[k].v, (short)0, acc01, false, false);
                acc10 = __builtin_amdgcn_wmma_f32_16x16x32_bf16(
                    false, A1.v, false, B0[k].v, (short)0, acc10, false, false);
                acc11 = __builtin_amdgcn_wmma_f32_16x16x32_bf16(
                    false, A1.v, false, B1[k].v, (short)0, acc11, false, false);
            }
        }

        // ---- fused epilogue: relu(h+b1)·w2, DPP butterfly over 16-lane N groups ----
#pragma unroll
        for (int r = 0; r < 8; ++r) {
            const int m = r + nhalf * 8;
            float a = acc00[r] + b1v0; a = a > 0.f ? a : 0.f;
            float b = acc01[r] + b1v1; b = b > 0.f ? b : 0.f;
            float v0 = a * w2v0 + b * w2v1;
            float c = acc10[r] + b1v0; c = c > 0.f ? c : 0.f;
            float d = acc11[r] + b1v1; d = d > 0.f ? d : 0.f;
            float v1 = c * w2v0 + d * w2v1;
            v0 = reduce16(v0);
            v1 = reduce16(v1);
            if (nlo == 0) {
                atomicAdd(&sOut[bi][m], v0);
                atomicAdd(&sOut[bi][16 + m], v1);
            }
        }
        __syncthreads();

        if (tid < 32) {
            const int e = tile * 32 + tid;
            if (e < total) out[e] = sOut[bi][tid] + b2v;
        }
        bi ^= 1;
    }
}

extern "C" void kernel_launch(void* const* d_in, const int* in_sizes, int n_in,
                              void* d_out, int out_size, void* d_ws, size_t ws_size,
                              hipStream_t stream) {
    const float* z_p = (const float*)d_in[0];
    const float* z_o = (const float*)d_in[1];
    const int*  idx0 = (const int*)d_in[2];
    const int*  idx1 = (const int*)d_in[3];
    const int*  idx2 = (const int*)d_in[4];
    const float* w1  = (const float*)d_in[5];
    const float* b1  = (const float*)d_in[6];
    const float* w2  = (const float*)d_in[7];
    const float* b2  = (const float*)d_in[8];
    float* out = (float*)d_out;

    const int E = in_sizes[2] / 2;
    const int total = 3 * E;
    const int numTiles = (total + 31) / 32;
    const int nzp = in_sizes[0];      // N_NODES * H
    const int nzo = in_sizes[1];

    // workspace layout: [w1t 64KB][zp_bf][zo_bf]
    const size_t need = 65536 + (size_t)(nzp + nzo) * 2;
    unsigned short* w1t  = (unsigned short*)d_ws;
    unsigned short* zp_bf = (unsigned short*)((char*)d_ws + 65536);
    unsigned short* zo_bf = zp_bf + nzp;
    const int use_bf16 = (ws_size >= need) ? 1 : 0;

    prep_w1t_kernel<<<(HID * 2 * HID + 255) / 256, 256, 0, stream>>>(w1, w1t);
    if (use_bf16) {
        cvt_bf16_kernel<<<(nzp / 4 + 255) / 256, 256, 0, stream>>>(z_p, zp_bf, nzp / 4);
        cvt_bf16_kernel<<<(nzo / 4 + 255) / 256, 256, 0, stream>>>(z_o, zo_bf, nzo / 4);
    }

    int blocks = numTiles < 2048 ? numTiles : 2048;
    edge_mlp_kernel<<<blocks, 128, 0, stream>>>(z_p, z_o, zp_bf, zo_bf,
                                                idx0, idx1, idx2,
                                                w1t, b1, w2, b2, out,
                                                E, numTiles, use_bf16);
}
